// EdgeDecoder_81484119539941
// MI455X (gfx1250) — matrix operand order repaired and checked by reading, not compiled, not run on hardware
//
#include <hip/hip_runtime.h>
#include <hip/hip_bf16.h>

// ---------------------------------------------------------------------------
// EdgeDecoder on MI455X (gfx1250):
//   out[e] = relu(concat(z[src[e]], z[dst[e]]) @ W1^T + b1) @ W2^T + b2
// E=500000, LATENT=256, HIDDEN=256, EDGE_DIM=128.
//
// bf16 WMMA (v_wmma_f32_16x16x32_bf16), f32 accumulation.
// z (102MB) is L2-resident (192MB L2). HBM traffic ~366MB => ~16us floor.
// This version removes the L2-bandwidth/latency bottleneck of round 1:
//  - W1/W2 pre-swizzled (kt-major) bf16 B-fragments, cooperatively staged
//    through LDS once per 8-wave block (8x less L2 traffic, ds-fed WMMA).
//  - Pipelined: next K-step's fragments prefetched into registers while
//    the current step's 16 WMMAs run; A-row loads issued before barriers.
//  - Output repacked via LDS into fully coalesced float4 stores.
// ---------------------------------------------------------------------------

typedef __attribute__((ext_vector_type(16))) __bf16 v16bf;
typedef __attribute__((ext_vector_type(8)))  float  v8f;

struct Frag32 { uint4 a; uint4 b; };

static __device__ __forceinline__ v16bf frag_cast(uint4 a, uint4 b) {
  Frag32 f{a, b};
  return __builtin_bit_cast(v16bf, f);
}

// f32 -> bf16, round-to-nearest-even
static __device__ __forceinline__ unsigned short f2bf(float x) {
  unsigned u = __builtin_bit_cast(unsigned, x);
  u += 0x7FFFu + ((u >> 16) & 1u);
  return (unsigned short)(u >> 16);
}

static __device__ __forceinline__ v16bf pack16(const float* f) {
  unsigned w[8];
#pragma unroll
  for (int i = 0; i < 8; ++i)
    w[i] = (unsigned)f2bf(f[2 * i]) | ((unsigned)f2bf(f[2 * i + 1]) << 16);
  return frag_cast(make_uint4(w[0], w[1], w[2], w[3]),
                   make_uint4(w[4], w[5], w[6], w[7]));
}

// ---------------------------------------------------------------------------
// Pre-swizzle W1 (256x512) and W2 (128x256) into bf16 WMMA B-fragments,
// *kt-major* so each K-step's fragments form one contiguous block:
//   wf1[((kt*16 + nt)*32 + lane)*16 + j],  k = kt*32 + 16*(lane>=16) + j,
//                                          n = nt*16 + (lane&15)
//   per-kt block: 16 frags * 32 lanes * 32B = 16 KB   (total 256 KB)
//   wf2[((kt*8  + nt)*32 + lane)*16 + j]   per-kt block: 8 KB (total 64 KB)
// ---------------------------------------------------------------------------
__global__ void prep_weights(const float* __restrict__ W1,
                             const float* __restrict__ W2,
                             unsigned short* __restrict__ wf1,
                             unsigned short* __restrict__ wf2) {
  int idx = blockIdx.x * 256 + threadIdx.x;
  if (idx < 131072) {
    int j    = idx & 15;
    int lane = (idx >> 4) & 31;
    int fidx = idx >> 9;
    int nt   = fidx & 15;
    int kt   = fidx >> 4;
    int n = nt * 16 + (lane & 15);
    int k = kt * 32 + ((lane >> 4) << 4) + j;
    wf1[idx] = f2bf(W1[n * 512 + k]);
  } else {
    int e = idx - 131072;
    if (e < 32768) {
      int j    = e & 15;
      int lane = (e >> 4) & 31;
      int fidx = e >> 9;
      int nt   = fidx & 7;
      int kt   = fidx >> 3;
      int n = nt * 16 + (lane & 15);
      int k = kt * 32 + ((lane >> 4) << 4) + j;
      wf2[e] = f2bf(W2[n * 256 + k]);
    }
  }
}

// ---------------------------------------------------------------------------
// Main kernel: 256 threads = 8 wave32 = 8 x 16-edge M-tiles per block.
// B fragments staged cooperatively in LDS (one load per block, 8 consumers).
// ---------------------------------------------------------------------------
__global__ __launch_bounds__(256) void edge_mlp_kernel(
    const float* __restrict__ z, const long long* __restrict__ ei,
    const float* __restrict__ b1, const float* __restrict__ b2,
    const unsigned short* __restrict__ wf1,
    const unsigned short* __restrict__ wf2,
    float* __restrict__ out, int E) {
  const int t    = threadIdx.x;
  const int lane = t & 31;
  const int wid  = t >> 5;
  const int m    = lane & 15;  // A-row / B,C,D-column owned by this lane
  const int hi   = lane >> 4;  // half-wave select

  const long long tiles = ((long long)E + 15) >> 4;
  const long long tile  = (long long)blockIdx.x * 8 + wid;
  const bool live = (tile < tiles);
  const long long e0 = live ? (tile << 4) : 0;

  long long e = e0 + m;
  if (e >= E) e = E - 1;
  const long long si = ei[e];
  const long long di = ei[(long long)E + e];
  const float* srow = z + si * 256;
  const float* drow = z + di * 256;

  __shared__ uint4          bstage[1024];       // 16 KB B-fragment staging
  __shared__ unsigned short hsh[8][16 * 264];   // per-wave h / out buffers
  unsigned short* hbuf = hsh[wid];

  // ---- Layer 1 accumulators seeded with b1 (C/D col = nt*16 + m) ----
  v8f acc[16];
#pragma unroll
  for (int nt = 0; nt < 16; ++nt) {
    float bv = b1[nt * 16 + m];
#pragma unroll
    for (int i = 0; i < 8; ++i) acc[nt][i] = bv;
  }

  const uint4* w1v = (const uint4*)wf1;
  // Preload staging registers for kt = 0 (64 B per thread).
  uint4 rr[4];
#pragma unroll
  for (int i = 0; i < 4; ++i) rr[i] = w1v[t * 4 + i];

  for (int kt = 0; kt < 16; ++kt) {
    // A-row loads for this kt issued early (in flight across the barriers).
    const float* base = (kt < 8) ? (srow + kt * 32) : (drow + (kt - 8) * 32);
    const float4* pa = (const float4*)(base + (hi ? 16 : 0));
    const float4* pb = (const float4*)(base + (hi ? 24 : 8));
    float4 q0 = pa[0], q1 = pa[1], q2 = pb[0], q3 = pb[1];

    __syncthreads();  // previous step's consumers done -> safe to overwrite
#pragma unroll
    for (int i = 0; i < 4; ++i) bstage[t * 4 + i] = rr[i];
    __syncthreads();  // staged fragments visible

    if (kt < 15) {    // prefetch next K-step's fragments into registers
#pragma unroll
      for (int i = 0; i < 4; ++i) rr[i] = w1v[(kt + 1) * 1024 + t * 4 + i];
    }
    if (kt < 14)      // warm L2/WGP$ two steps ahead
      __builtin_prefetch(w1v + (kt + 2) * 1024 + t * 4, 0, 1);

    float fa[16] = {q0.x, q0.y, q0.z, q0.w, q1.x, q1.y, q1.z, q1.w,
                    q2.x, q2.y, q2.z, q2.w, q3.x, q3.y, q3.z, q3.w};
    v16bf afrag = pack16(fa);
#pragma unroll
    for (int nt = 0; nt < 16; ++nt) {
      int bi = (nt * 32 + lane) * 2;
      v16bf bfrag = frag_cast(bstage[bi], bstage[bi + 1]);
      acc[nt] = __builtin_amdgcn_wmma_f32_16x16x32_bf16(
          false, afrag, false, bfrag, (short)0, acc[nt], false, false);
    }
  }

  // ---- ReLU + C-layout -> per-wave LDS h (row-major bf16, padded 264) ----
#pragma unroll
  for (int nt = 0; nt < 16; ++nt) {
    int col = nt * 16 + m;
#pragma unroll
    for (int r = 0; r < 8; ++r) {
      float v = acc[nt][r];
      v = v > 0.f ? v : 0.f;
      hbuf[(r + 8 * hi) * 264 + col] = f2bf(v);
    }
  }

  // ---- Layer 2: acc2 seeded with b2 ----
  v8f acc2[8];
#pragma unroll
  for (int nt = 0; nt < 8; ++nt) {
    float bv = b2[nt * 16 + m];
#pragma unroll
    for (int i = 0; i < 8; ++i) acc2[nt][i] = bv;
  }

  const uint4* w2v = (const uint4*)wf2;
  uint4 rr2[2];
#pragma unroll
  for (int i = 0; i < 2; ++i) rr2[i] = w2v[t * 2 + i];

  for (int kt = 0; kt < 8; ++kt) {
    __syncthreads();
#pragma unroll
    for (int i = 0; i < 2; ++i) bstage[t * 2 + i] = rr2[i];
    __syncthreads();
    if (kt < 7) {
#pragma unroll
      for (int i = 0; i < 2; ++i) rr2[i] = w2v[(kt + 1) * 512 + t * 2 + i];
    }

    // A fragment of h from per-wave LDS: two 16B-aligned runs of 8 bf16.
    const uint4* ha = (const uint4*)(hbuf + m * 264 + kt * 32 + hi * 16);
    const uint4* hb = (const uint4*)(hbuf + m * 264 + kt * 32 + 8 + hi * 16);
    v16bf afrag = frag_cast(ha[0], hb[0]);
#pragma unroll
    for (int nt = 0; nt < 8; ++nt) {
      int bi = (nt * 32 + lane) * 2;
      v16bf bfrag = frag_cast(bstage[bi], bstage[bi + 1]);
      acc2[nt] = __builtin_amdgcn_wmma_f32_16x16x32_bf16(
          false, afrag, false, bfrag, (short)0, acc2[nt], false, false);
    }
  }

  // ---- Output: repack through per-wave LDS, coalesced float4 stores ----
  // Reuse hbuf as [16][132] f32 (16*132*4 = 8448 B = exactly the h region).
  float* obuf = (float*)hbuf;
#pragma unroll
  for (int nt = 0; nt < 8; ++nt) {
    int col = nt * 16 + m;
#pragma unroll
    for (int r = 0; r < 8; ++r)
      obuf[(r + 8 * hi) * 132 + col] = acc2[nt][r];
  }
  // (same-wave ds_store -> ds_load ordering; compiler inserts DScnt waits)
  float* outb = out + e0 * 128;
#pragma unroll
  for (int r2 = 0; r2 < 16; ++r2) {
    if (live && (e0 + r2 < (long long)E)) {
      const float4* s4 = (const float4*)(obuf + r2 * 132 + lane * 4);
      ((float4*)(outb + r2 * 128))[lane] = s4[0];
    }
  }
}

extern "C" void kernel_launch(void* const* d_in, const int* in_sizes, int n_in,
                              void* d_out, int out_size, void* d_ws,
                              size_t ws_size, hipStream_t stream) {
  const float*     z  = (const float*)d_in[0];
  const long long* ei = (const long long*)d_in[1];  // int64 edge_index (2,E)
  const float*     W1 = (const float*)d_in[2];
  const float*     b1 = (const float*)d_in[3];
  const float*     W2 = (const float*)d_in[4];
  const float*     b2 = (const float*)d_in[5];
  float* out = (float*)d_out;

  const int E = in_sizes[1] / 2;

  unsigned short* wf1 = (unsigned short*)d_ws;   // 131072 bf16 (256 KB)
  unsigned short* wf2 = wf1 + 131072;            //  32768 bf16 ( 64 KB)

  prep_weights<<<640, 256, 0, stream>>>(W1, W2, wf1, wf2);

  const long long tiles  = ((long long)E + 15) >> 4;
  const int       blocks = (int)((tiles + 7) >> 3);
  edge_mlp_kernel<<<blocks, 256, 0, stream>>>(z, ei, b1, b2, wf1, wf2, out, E);
}
